// AdaptiveGraphTopology_46385646797016
// MI455X (gfx1250) — matrix operand order repaired and checked by reading, compiled, not run on hardware
//
#include <hip/hip_runtime.h>

#define N_NODES 1024
#define HDIM    128

typedef __attribute__((ext_vector_type(16))) _Float16 v16h;
typedef __attribute__((ext_vector_type(8)))  float    v8f;
typedef __attribute__((ext_vector_type(4)))  int      v4i;

union Frag {
    v16h f;      // WMMA operand: 16 halves per lane (8 VGPRs)
    v4i  i4[2];  // two 16-byte load chunks
};

// ---------------------------------------------------------------------------
// Phase 1: a[i,n] = emb[i,:]·W1[n,0:128], b[i,n] = emb[i,:]·W1[n,128:256]+b1[n]
// Stored as f16 for the WMMA stage. Tiny cost (~67 MFLOP vs ~4.3 GFLOP of WMMA).
// ---------------------------------------------------------------------------
__global__ __launch_bounds__(HDIM)
void agt_prep_kernel(const float* __restrict__ emb, const float* __restrict__ W1,
                     const float* __restrict__ b1,  const float* __restrict__ W2,
                     _Float16* __restrict__ aOut, _Float16* __restrict__ bOut,
                     _Float16* __restrict__ w2Out) {
    const int i = blockIdx.x;    // node index
    const int n = threadIdx.x;   // output feature
    __shared__ float er[HDIM];
    er[n] = emb[i * HDIM + n];
    __syncthreads();

    const float* w1row = W1 + n * (2 * HDIM);
    float sa = 0.f, sb = 0.f;
#pragma unroll 8
    for (int k = 0; k < HDIM; ++k) {
        float e = er[k];
        sa = fmaf(e, w1row[k], sa);
        sb = fmaf(e, w1row[HDIM + k], sb);
    }
    aOut[i * HDIM + n] = (_Float16)sa;
    bOut[i * HDIM + n] = (_Float16)(sb + b1[n]);   // fold b1 into b
    if (i == 0) w2Out[n] = (_Float16)W2[n];
}

// ---------------------------------------------------------------------------
// Phase 2: per wave, one 16x16 (i,j) output tile.
//   A-frag (16x32 f16): rows = 16 j's, K = h-block; built as relu(a_f16+b_f16)
//     with packed v_pk_add_f16 / v_pk_max_num_f16 on the VALU.
//   B-frag (32x16 f16): w2 replicated into all 16 columns -> every column of
//     the f32 accumulator equals the score vector for the current i.
//   v_wmma_f32_16x16x32_f16 (XDL) co-executes with the packed VALU ops.
// ---------------------------------------------------------------------------
__global__ __launch_bounds__(256)
void agt_score_kernel(const _Float16* __restrict__ aF, const _Float16* __restrict__ bF,
                      const _Float16* __restrict__ w2F, const float* __restrict__ b2,
                      float* __restrict__ wOut, float* __restrict__ mOut) {
    __shared__ float tile[8][256];            // one 16x16 f32 tile per wave

    const int wid    = threadIdx.x >> 5;
    const int lane   = threadIdx.x & 31;
    const int tileId = blockIdx.x * 8 + wid;  // 4096 tiles = 64x64
    const int jt     = (tileId & 63) << 4;    // j tile base
    const int it     = (tileId >> 6) << 4;    // i tile base
    const int jl     = lane & 15;             // A-matrix row owned by this lane
    const int hiHalf = lane >> 4;             // lane group selects K sub-range
    const float b2v  = b2[0];

    // B fragments: B[k,n] = w2[kb*32 + k] for all n. Lane group g holds
    // K=8g..8g+7 and K=16+8g..16+8g+7 of the block (2 K's per VGPR).
    Frag wf[4];
#pragma unroll
    for (int kb = 0; kb < 4; ++kb) {
        const v4i* p = (const v4i*)(w2F + kb * 32 + hiHalf * 8);
        wf[kb].i4[0] = p[0];
        wf[kb].i4[1] = p[2];
    }

    // b fragments for row j = jt + jl, resident across the whole i loop.
    // A 16-bit 16x32 layout: lane holds halves K = c0..c0+7 and c0+16..c0+23,
    // c0 = kb*32 + 8*laneGroup  -> two contiguous 16-byte chunks per block.
    Frag bf[4];
#pragma unroll
    for (int kb = 0; kb < 4; ++kb) {
        const _Float16* base = bF + (jt + jl) * HDIM + kb * 32 + hiHalf * 8;
        bf[kb].i4[0] = *(const v4i*)(base);
        bf[kb].i4[1] = *(const v4i*)(base + 16);
    }

    const v16h z16 = (v16h)(_Float16)0;

    for (int il = 0; il < 16; ++il) {
        const int i = it + il;
        v8f acc = {};
#pragma unroll
        for (int kb = 0; kb < 4; ++kb) {
            Frag h;
            const _Float16* abase = aF + i * HDIM + kb * 32 + hiHalf * 8;
            h.i4[0] = *(const v4i*)(abase);
            h.i4[1] = *(const v4i*)(abase + 16);
            // hidden = relu(a + b'): 8x v_pk_add_f16 + 8x v_pk_max_num_f16
            h.f = __builtin_elementwise_max(h.f + bf[kb].f, z16);
            acc = __builtin_amdgcn_wmma_f32_16x16x32_f16(
                      false, h.f, false, wf[kb].f, (short)0, acc, false, false);
        }
        // D layout: lane 0 VGPR r = row j=r ; lane 16 VGPR r = row j=r+8.
        if (jl == 0) {
            float* dst = &tile[wid][il * 16 + hiHalf * 8];
#pragma unroll
            for (int r = 0; r < 8; ++r) dst[r] = acc[r];
        }
    }
    __syncthreads();

    // Epilogue: 8 elements per lane — sigmoid, diagonal zero, mask.
#pragma unroll
    for (int k = 0; k < 8; ++k) {
        const int e   = lane * 8 + k;
        const int il  = e >> 4, jl2 = e & 15;
        const float s = tile[wid][e] + b2v;
        const int gi  = it + il, gj = jt + jl2;
        const bool eye = (gi == gj);
        const float wgt = 1.f / (1.f + __expf(-s));
        wOut[gi * N_NODES + gj] = eye ? 0.f : wgt;
        mOut[gi * N_NODES + gj] = (!eye && s > 0.f) ? 1.f : 0.f;  // sig(s)>0.5 <=> s>0
    }
}

extern "C" void kernel_launch(void* const* d_in, const int* in_sizes, int n_in,
                              void* d_out, int out_size, void* d_ws, size_t ws_size,
                              hipStream_t stream) {
    (void)in_sizes; (void)n_in; (void)out_size; (void)ws_size;
    const float* emb = (const float*)d_in[0];   // (1024,128)
    const float* W1  = (const float*)d_in[1];   // (128,256)
    const float* b1  = (const float*)d_in[2];   // (128,)
    const float* W2  = (const float*)d_in[3];   // (1,128)
    const float* b2  = (const float*)d_in[4];   // (1,)
    // d_in[5] = temperature: unused by the reference math.

    _Float16* aF  = (_Float16*)d_ws;            // 1024*128 halves = 256 KB
    _Float16* bFp = aF + N_NODES * HDIM;        // 256 KB
    _Float16* w2F = bFp + N_NODES * HDIM;       // 256 B

    float* wOut = (float*)d_out;                // weights (1024*1024)
    float* mOut = wOut + N_NODES * N_NODES;     // mask as 0/1 floats

    agt_prep_kernel<<<N_NODES, HDIM, 0, stream>>>(emb, W1, b1, W2, aF, bFp, w2F);
    agt_score_kernel<<<(64 * 64) / 8, 256, 0, stream>>>(aF, bFp, w2F, b2, wOut, mOut);
}
